// RelHetGraph_19284403159435
// MI455X (gfx1250) — matrix operand order repaired and checked by reading, not compiled, not run on hardware
//
#include <hip/hip_runtime.h>

// ---------------------------------------------------------------------------
// CDNA5 (gfx1250) RelHetGraph: bf16-end-to-end WMMA GEMMs.
// All GEMM operands live in global memory as bf16 (activations converted once,
// weights transposed+converted once), so tile staging is pure b128 copies and
// the inner loop is ds_load_b128 + v_wmma_f32_16x16x32_bf16.
// Edge softmax/aggregation: int-encoded atomicMax + float atomicAdd (L2-resident).
// ---------------------------------------------------------------------------

#define DIN     768
#define TILE_M  128
#define TILE_N  64
#define TILE_K  32

typedef __attribute__((ext_vector_type(16))) __bf16       v16bf;
typedef __attribute__((ext_vector_type(8)))  float        v8f;
typedef __attribute__((ext_vector_type(8)))  unsigned int v8u;

__device__ __forceinline__ unsigned short f2bf(float f) {
  unsigned int u = __builtin_bit_cast(unsigned int, f);
  u += 0x7FFFu + ((u >> 16) & 1u);          // round-to-nearest-even
  return (unsigned short)(u >> 16);
}

// monotonic float <-> int encoding for atomicMax on floats
__device__ __forceinline__ int enc_f(float v) {
  int i = __builtin_bit_cast(int, v);
  return i >= 0 ? i : (i ^ 0x7fffffff);
}
__device__ __forceinline__ float dec_f(int i) {
  int u = i >= 0 ? i : (i ^ 0x7fffffff);
  return __builtin_bit_cast(float, u);
}

// ---------------------------------------------------------------------------
// WMMA GEMM (TN):  C[M,N] = act(A[M,K] @ Bt[N,K]^T + bias)
// A: bf16 row-major [M,K]; Bt: bf16 [N,K] (pre-transposed weights).
// K % 32 == 0, N % 64 == 0; M guarded. 256 thr = 8 waves; block tile 128x64.
// Output: f32 (Cf) or bf16 (Cb) -- exactly one non-null.
// ---------------------------------------------------------------------------
__global__ __launch_bounds__(256)
void gemm_bf16_tn(const unsigned short* __restrict__ A,
                  const unsigned short* __restrict__ Bt,
                  const float* __restrict__ bias,
                  float* __restrict__ Cf, unsigned short* __restrict__ Cb,
                  int M, int N, int K, int do_relu)
{
  __shared__ unsigned short As[TILE_M * TILE_K];  // [128][32] (m,k)
  __shared__ unsigned short Bs[TILE_N * TILE_K];  // [64][32]  (n,k)

  const int tid  = threadIdx.x;
  const int lane = tid & 31;
  const int wave = tid >> 5;
  const int wm   = wave >> 1;            // 0..3 rows
  const int wn   = wave & 1;             // 0..1 cols
  const int bm   = blockIdx.x * TILE_M;
  const int bn   = blockIdx.y * TILE_N;
  const int lm   = lane & 15;
  const int lh   = lane >> 4;

  v8f z = {};
  v8f acc00 = z, acc01 = z, acc10 = z, acc11 = z;

  for (int k0 = 0; k0 < K; k0 += TILE_K) {
    // ---- stage A tile: 128x32 bf16 = 512 x 16B, 2 vectors/thread ----
#pragma unroll
    for (int i = 0; i < 2; ++i) {
      int idx  = tid + i * 256;          // 0..511
      int r    = idx >> 2;               // 0..127
      int kv   = (idx & 3) << 3;         // 0,8,16,24
      int grow = bm + r;
      uint4 v = make_uint4(0u, 0u, 0u, 0u);
      if (grow < M) {
        const unsigned short* ap = A + (size_t)grow * K + k0 + kv;
        v = *reinterpret_cast<const uint4*>(ap);
        if (k0 + TILE_K < K) __builtin_prefetch(ap + TILE_K, 0, 1);
      }
      *reinterpret_cast<uint4*>(&As[r * TILE_K + kv]) = v;
    }
    // ---- stage B tile: 64x32 bf16 = 256 x 16B, 1 vector/thread ----
    {
      int n  = tid >> 2;                 // 0..63
      int kv = (tid & 3) << 3;           // 0,8,16,24
      const unsigned short* bp = Bt + (size_t)(bn + n) * K + k0 + kv;
      uint4 v = *reinterpret_cast<const uint4*>(bp);
      if (k0 + TILE_K < K) __builtin_prefetch(bp + TILE_K, 0, 1);
      *reinterpret_cast<uint4*>(&Bs[n * TILE_K + kv]) = v;
    }
    __syncthreads();

    // ---- fragments per ISA layout ----
    v16bf af0, af1, bf0, bf1;
    {
      int m = wm * 32 + lm;
      v8u u;
#pragma unroll
      for (int j = 0; j < 8; ++j) {
        int kp = (j < 4 ? 2 * j : 2 * j + 8) + lh * 8;
        u[j] = *reinterpret_cast<const unsigned int*>(&As[m * TILE_K + kp]);
      }
      af0 = __builtin_bit_cast(v16bf, u);
      m = wm * 32 + 16 + lm;
#pragma unroll
      for (int j = 0; j < 8; ++j) {
        int kp = (j < 4 ? 2 * j : 2 * j + 8) + lh * 8;
        u[j] = *reinterpret_cast<const unsigned int*>(&As[m * TILE_K + kp]);
      }
      af1 = __builtin_bit_cast(v16bf, u);
    }
    {
      int n = wn * 32 + lm;
      v8u u;
#pragma unroll
      for (int j = 0; j < 8; ++j) {
        int kp = (j < 4 ? 2 * j : 2 * j + 8) + lh * 8;
        u[j] = *reinterpret_cast<const unsigned int*>(&Bs[n * TILE_K + kp]);
      }
      bf0 = __builtin_bit_cast(v16bf, u);
      n = wn * 32 + 16 + lm;
#pragma unroll
      for (int j = 0; j < 8; ++j) {
        int kp = (j < 4 ? 2 * j : 2 * j + 8) + lh * 8;
        u[j] = *reinterpret_cast<const unsigned int*>(&Bs[n * TILE_K + kp]);
      }
      bf1 = __builtin_bit_cast(v16bf, u);
    }

    acc00 = __builtin_amdgcn_wmma_f32_16x16x32_bf16(false, af0, false, bf0, (short)0, acc00, false, false);
    acc01 = __builtin_amdgcn_wmma_f32_16x16x32_bf16(false, af0, false, bf1, (short)0, acc01, false, false);
    acc10 = __builtin_amdgcn_wmma_f32_16x16x32_bf16(false, af1, false, bf0, (short)0, acc10, false, false);
    acc11 = __builtin_amdgcn_wmma_f32_16x16x32_bf16(false, af1, false, bf1, (short)0, acc11, false, false);
    __syncthreads();
  }

  // ---- epilogue: C layout -> M = r + 8*lh, N = lane&15 ----
  v8f accs[2][2] = {{acc00, acc01}, {acc10, acc11}};
#pragma unroll
  for (int sm = 0; sm < 2; ++sm)
#pragma unroll
    for (int sn = 0; sn < 2; ++sn) {
      int col = bn + wn * 32 + sn * 16 + lm;
      float bv = bias ? bias[col] : 0.f;
#pragma unroll
      for (int r = 0; r < 8; ++r) {
        int row = bm + wm * 32 + sm * 16 + r + lh * 8;
        if (row < M) {
          float v = accs[sm][sn][r] + bv;
          if (do_relu) v = fmaxf(v, 0.f);
          if (Cf) Cf[(size_t)row * N + col] = v;
          else    Cb[(size_t)row * N + col] = f2bf(v);
        }
      }
    }
}

// ---------------------------------------------------------------------------
// One-time conversions
// ---------------------------------------------------------------------------
// fp32 -> bf16, 4 elements/thread (n % 4 == 0 for all users)
__global__ void k_cvt_bf16(const float* __restrict__ X, unsigned short* __restrict__ Y, long n4) {
  long i = (long)blockIdx.x * 256 + threadIdx.x;
  if (i >= n4) return;
  float4 v = reinterpret_cast<const float4*>(X)[i];
  unsigned int lo = (unsigned int)f2bf(v.x) | ((unsigned int)f2bf(v.y) << 16);
  unsigned int hi = (unsigned int)f2bf(v.z) | ((unsigned int)f2bf(v.w) << 16);
  reinterpret_cast<uint2*>(Y)[i] = make_uint2(lo, hi);
}

// W[K,N] f32 -> Wt[N,K] bf16 (tiled LDS transpose)
__global__ __launch_bounds__(256)
void k_transpose_bf16(const float* __restrict__ W, unsigned short* __restrict__ Wt,
                      int K, int N) {
  __shared__ unsigned short t[32][33];
  int bx = blockIdx.x * 32;  // N
  int by = blockIdx.y * 32;  // K
  int tx = threadIdx.x & 31, ty = threadIdx.x >> 5;  // ty 0..7
#pragma unroll
  for (int i = 0; i < 32; i += 8) {
    int k = by + ty + i, n = bx + tx;
    float v = (k < K && n < N) ? W[(size_t)k * N + n] : 0.f;
    t[ty + i][tx] = f2bf(v);
  }
  __syncthreads();
#pragma unroll
  for (int i = 0; i < 32; i += 8) {
    int n = bx + ty + i, k = by + tx;
    if (n < N && k < K) Wt[(size_t)n * K + k] = t[tx][ty + i];
  }
}

// ---------------------------------------------------------------------------
// Utility / sparse kernels
// ---------------------------------------------------------------------------
__global__ void k_fill_f32(float* p, long n, float v) {
  long i = (long)blockIdx.x * 256 + threadIdx.x;
  if (i < n) p[i] = v;
}
__global__ void k_fill_i32(int* p, long n, int v) {
  long i = (long)blockIdx.x * 256 + threadIdx.x;
  if (i < n) p[i] = v;
}
__global__ void k_add_bias_rows(float* X, const float* __restrict__ b, long n, int D) {
  long i = (long)blockIdx.x * 256 + threadIdx.x;
  if (i < n) X[i] += b[(int)(i % D)];
}

// out[n*H+h] = dot(G[n, h*C : (h+1)*C], a[h, :])
__global__ void k_attn_score(const float* __restrict__ G, const float* __restrict__ a,
                             float* __restrict__ out, int Nn, int H, int C) {
  int id = blockIdx.x * 256 + threadIdx.x;
  if (id >= Nn * H) return;
  int n = id / H, h = id % H;
  const float* g  = G + (size_t)n * H * C + (size_t)h * C;
  const float* av = a + (size_t)h * C;
  float s = 0.f;
  for (int c = 0; c < C; ++c) s += g[c] * av[c];
  out[id] = s;
}

__global__ void k_edge_max(const int* __restrict__ src, const int* __restrict__ dst,
                           const float* __restrict__ als, const float* __restrict__ ald,
                           int* __restrict__ mx, int E, int H) {
  int e = blockIdx.x * 256 + threadIdx.x;
  if (e >= E) return;
  int s = src[e], d = dst[e];
  for (int h = 0; h < H; ++h) {
    float v = als[s * H + h] + ald[d * H + h];
    v = v > 0.f ? v : 0.2f * v;
    atomicMax(&mx[d * H + h], enc_f(v));
  }
}

__global__ void k_edge_exp(const int* __restrict__ src, const int* __restrict__ dst,
                           const float* __restrict__ als, const float* __restrict__ ald,
                           const int* __restrict__ mx, float* __restrict__ den,
                           float* __restrict__ exb, int E, int H) {
  int e = blockIdx.x * 256 + threadIdx.x;
  if (e >= E) return;
  int s = src[e], d = dst[e];
  for (int h = 0; h < H; ++h) {
    float v = als[s * H + h] + ald[d * H + h];
    v = v > 0.f ? v : 0.2f * v;
    float m  = dec_f(mx[d * H + h]);
    float ex = __expf(v - m);
    exb[e * H + h] = ex;
    atomicAdd(&den[d * H + h], ex);
  }
}

// one block per edge, HC threads; accum[dst, c] += G[src, c] * alpha
__global__ void k_edge_scatter(const int* __restrict__ src, const int* __restrict__ dst,
                               const float* __restrict__ exb, const float* __restrict__ den,
                               const float* __restrict__ G, float* __restrict__ out,
                               int H, int C) {
  int e = blockIdx.x, c = threadIdx.x;
  int HC = H * C;
  int s = src[e], d = dst[e];
  int h = c / C;
  float alpha = exb[e * H + h] / (den[d * H + h] + 1e-16f);
  atomicAdd(&out[(size_t)d * HC + c], G[(size_t)s * HC + c] * alpha);
}

// one block (256 threads) per row, D == 256; optional f32 and/or bf16 outputs
__global__ __launch_bounds__(256)
void k_relu_ln(const float* __restrict__ X, float* __restrict__ Yf,
               unsigned short* __restrict__ Yb, int D, int do_relu) {
  int row = blockIdx.x, tid = threadIdx.x;
  __shared__ float red[256];
  float v = X[(size_t)row * D + tid];
  if (do_relu) v = fmaxf(v, 0.f);
  red[tid] = v; __syncthreads();
  for (int s = 128; s > 0; s >>= 1) { if (tid < s) red[tid] += red[tid + s]; __syncthreads(); }
  float mu = red[0] / (float)D; __syncthreads();
  float dv = v - mu;
  red[tid] = dv * dv; __syncthreads();
  for (int s = 128; s > 0; s >>= 1) { if (tid < s) red[tid] += red[tid + s]; __syncthreads(); }
  float var = red[0] / (float)D;
  float o = dv * rsqrtf(var + 1e-5f);
  size_t idx = (size_t)row * D + tid;
  if (Yf) Yf[idx] = o;
  if (Yb) Yb[idx] = f2bf(o);
}

// ---------------------------------------------------------------------------
static inline int cdiv_i(long a, long b) { return (int)((a + b - 1) / b); }

extern "C" void kernel_launch(void* const* d_in, const int* in_sizes, int n_in,
                              void* d_out, int out_size, void* d_ws, size_t ws_size,
                              hipStream_t stream) {
  (void)n_in; (void)out_size; (void)ws_size;

  const float* x_s = (const float*)d_in[0];
  const float* x_w = (const float*)d_in[1];
  const float* x_d = (const float*)d_in[2];
  const int* ei_ss = (const int*)d_in[3];
  const int* ei_sa = (const int*)d_in[4];
  const int* ei_ws = (const int*)d_in[5];
  const int* ei_sw = (const int*)d_in[6];
  const int* ei_ds = (const int*)d_in[7];
  const int* ei_sd = (const int*)d_in[8];
  const float* Wls = (const float*)d_in[9];
  const float* bls = (const float*)d_in[10];
  const float* Wlw = (const float*)d_in[11];
  const float* blw = (const float*)d_in[12];
  const float* Wld = (const float*)d_in[13];
  const float* bld = (const float*)d_in[14];
  const float* W1s = (const float*)d_in[15];
  const float* W1d = (const float*)d_in[16];
  const float* a1s = (const float*)d_in[17];
  const float* a1d = (const float*)d_in[18];
  const float* b1  = (const float*)d_in[19];
  const float* W2s = (const float*)d_in[20];
  const float* W2d = (const float*)d_in[21];
  const float* a2s = (const float*)d_in[22];
  const float* a2d = (const float*)d_in[23];
  const float* b2  = (const float*)d_in[24];
  const float* Wp1 = (const float*)d_in[25];
  const float* bp1 = (const float*)d_in[26];
  const float* Wp2 = (const float*)d_in[27];
  const float* bp2 = (const float*)d_in[28];

  const int NS  = in_sizes[0] / DIN;
  const int NW  = in_sizes[1] / DIN;
  const int ND  = in_sizes[2] / DIN;
  const int ESS = in_sizes[3] / 2, ESA = in_sizes[4] / 2, EWS = in_sizes[5] / 2;
  const int ESW = in_sizes[6] / 2, EDS = in_sizes[7] / 2, ESD = in_sizes[8] / 2;
  const int HC = 256;

  // ---- workspace carve (bytes, 256B aligned) ----
  char* base = (char*)d_ws;
  size_t off = 0;
  auto walloc = [&](size_t bytes) {
    void* p = base + off;
    off += (bytes + 255) & ~(size_t)255;
    return p;
  };
  typedef unsigned short ush;
  // bf16 activations
  ush* xs_bf = (ush*)walloc((size_t)NS * DIN * 2);
  ush* xw_bf = (ush*)walloc((size_t)NW * DIN * 2);
  ush* xd_bf = (ush*)walloc((size_t)ND * DIN * 2);
  ush* hs_bf = (ush*)walloc((size_t)NS * DIN * 2);
  ush* hw_bf = (ush*)walloc((size_t)NW * DIN * 2);
  ush* hd_bf = (ush*)walloc((size_t)ND * DIN * 2);
  ush* s1_bf = (ush*)walloc((size_t)NS * HC * 2);
  ush* w1_bf = (ush*)walloc((size_t)NW * HC * 2);
  ush* d1_bf = (ush*)walloc((size_t)ND * HC * 2);
  ush* s2_bf = (ush*)walloc((size_t)NS * HC * 2);
  ush* p1_bf = (ush*)walloc((size_t)NS * 128 * 2);
  // transposed bf16 weights
  ush* WlsT = (ush*)walloc((size_t)DIN * DIN * 2);
  ush* WlwT = (ush*)walloc((size_t)DIN * DIN * 2);
  ush* WldT = (ush*)walloc((size_t)DIN * DIN * 2);
  ush* W1sT = (ush*)walloc((size_t)6 * HC * DIN * 2);
  ush* W1dT = (ush*)walloc((size_t)6 * HC * DIN * 2);
  ush* W2sT = (ush*)walloc((size_t)6 * HC * HC * 2);
  ush* W2dT = (ush*)walloc((size_t)6 * HC * HC * 2);
  ush* Wp1T = (ush*)walloc((size_t)128 * HC * 2);
  ush* Wp2T = (ush*)walloc((size_t)128 * 128 * 2);
  // f32 buffers
  float* S1   = (float*)walloc((size_t)NS * HC * 4);
  float* W1b  = (float*)walloc((size_t)NW * HC * 4);
  float* D1   = (float*)walloc((size_t)ND * HC * 4);
  float* S2   = (float*)walloc((size_t)NS * HC * 4);
  float* tmpS = (float*)walloc((size_t)NW * HC * 4);
  float* tmpD = (float*)walloc((size_t)NW * HC * 4);
  float* alsb = (float*)walloc((size_t)NW * 4 * 4);
  float* aldb = (float*)walloc((size_t)NW * 4 * 4);
  int*   mxb  = (int*)walloc((size_t)NW * 4 * 4);
  float* denb = (float*)walloc((size_t)NW * 4 * 4);
  float* exb  = (float*)walloc((size_t)200000 * 4 * 4);

  auto gemm_f = [&](const ush* A, const ush* Bt, const float* bias, float* C,
                    int M, int N, int K, int relu) {
    dim3 g(cdiv_i(M, TILE_M), N / TILE_N);
    gemm_bf16_tn<<<g, 256, 0, stream>>>(A, Bt, bias, C, nullptr, M, N, K, relu);
  };
  auto gemm_b = [&](const ush* A, const ush* Bt, const float* bias, ush* C,
                    int M, int N, int K, int relu) {
    dim3 g(cdiv_i(M, TILE_M), N / TILE_N);
    gemm_bf16_tn<<<g, 256, 0, stream>>>(A, Bt, bias, nullptr, C, M, N, K, relu);
  };
  auto cvt = [&](const float* X, ush* Y, size_t n) {
    k_cvt_bf16<<<cdiv_i((long)(n / 4), 256), 256, 0, stream>>>(X, Y, (long)(n / 4));
  };
  auto transp = [&](const float* W, ush* Wt, int K, int N) {
    dim3 g(cdiv_i(N, 32), cdiv_i(K, 32));
    k_transpose_bf16<<<g, 256, 0, stream>>>(W, Wt, K, N);
  };
  auto fillf = [&](float* p, size_t n, float v) {
    k_fill_f32<<<cdiv_i((long)n, 256), 256, 0, stream>>>(p, (long)n, v);
  };
  auto filli = [&](int* p, size_t n, int v) {
    k_fill_i32<<<cdiv_i((long)n, 256), 256, 0, stream>>>(p, (long)n, v);
  };

  // ---- one-time conversions ----
  cvt(x_s, xs_bf, (size_t)NS * DIN);
  cvt(x_w, xw_bf, (size_t)NW * DIN);
  cvt(x_d, xd_bf, (size_t)ND * DIN);
  transp(Wls, WlsT, DIN, DIN);
  transp(Wlw, WlwT, DIN, DIN);
  transp(Wld, WldT, DIN, DIN);
  for (int t = 0; t < 6; ++t) {
    transp(W1s + (size_t)t * DIN * HC, W1sT + (size_t)t * HC * DIN, DIN, HC);
    transp(W1d + (size_t)t * DIN * HC, W1dT + (size_t)t * HC * DIN, DIN, HC);
    transp(W2s + (size_t)t * HC * HC,  W2sT + (size_t)t * HC * HC,  HC,  HC);
    transp(W2d + (size_t)t * HC * HC,  W2dT + (size_t)t * HC * HC,  HC,  HC);
  }
  transp(Wp1, Wp1T, HC, 128);
  transp(Wp2, Wp2T, 128, 128);

  // ---- stage 0: input linears + relu (bf16 out; only consumed by GEMMs) ----
  gemm_b(xs_bf, WlsT, bls, hs_bf, NS, DIN, DIN, 1);
  gemm_b(xw_bf, WlwT, blw, hw_bf, NW, DIN, DIN, 1);
  gemm_b(xd_bf, WldT, bld, hd_bf, ND, DIN, DIN, 1);

  // ---- one GAT edge-type pass, accumulating into `accum` ----
  auto gat = [&](const ush* Xsrc, int Ns, const ush* Xdst, int Nd,
                 const int* ei, int E,
                 const ush* WsT, const ush* WdT, const float* as_, const float* ad_,
                 int Din_, int H, int C, float* accum) {
    int hc = H * C;  // == 256 for both conv layers
    gemm_f(Xsrc, WsT, nullptr, tmpS, Ns, hc, Din_, 0);
    gemm_f(Xdst, WdT, nullptr, tmpD, Nd, hc, Din_, 0);
    k_attn_score<<<cdiv_i((long)Ns * H, 256), 256, 0, stream>>>(tmpS, as_, alsb, Ns, H, C);
    k_attn_score<<<cdiv_i((long)Nd * H, 256), 256, 0, stream>>>(tmpD, ad_, aldb, Nd, H, C);
    filli(mxb, (size_t)Nd * H, (int)0x80000000);
    fillf(denb, (size_t)Nd * H, 0.f);
    k_edge_max<<<cdiv_i(E, 256), 256, 0, stream>>>(ei, ei + E, alsb, aldb, mxb, E, H);
    k_edge_exp<<<cdiv_i(E, 256), 256, 0, stream>>>(ei, ei + E, alsb, aldb, mxb, denb, exb, E, H);
    k_edge_scatter<<<E, hc, 0, stream>>>(ei, ei + E, exb, denb, tmpS, accum, H, C);
  };

  // ---- conv1 (H=4, C=64) ----
  fillf(S1, (size_t)NS * HC, 0.f);
  fillf(W1b, (size_t)NW * HC, 0.f);
  fillf(D1, (size_t)ND * HC, 0.f);
  const size_t w1t = (size_t)HC * DIN, a1st = 256, bst = 256;
  gat(hs_bf, NS, hs_bf, NS, ei_ss, ESS, W1sT + 0 * w1t, W1dT + 0 * w1t, a1s + 0 * a1st, a1d + 0 * a1st, DIN, 4, 64, S1);
  gat(hs_bf, NS, hs_bf, NS, ei_sa, ESA, W1sT + 1 * w1t, W1dT + 1 * w1t, a1s + 1 * a1st, a1d + 1 * a1st, DIN, 4, 64, S1);
  gat(hw_bf, NW, hs_bf, NS, ei_ws, EWS, W1sT + 2 * w1t, W1dT + 2 * w1t, a1s + 2 * a1st, a1d + 2 * a1st, DIN, 4, 64, S1);
  gat(hd_bf, ND, hs_bf, NS, ei_ds, EDS, W1sT + 4 * w1t, W1dT + 4 * w1t, a1s + 4 * a1st, a1d + 4 * a1st, DIN, 4, 64, S1);
  gat(hs_bf, NS, hw_bf, NW, ei_sw, ESW, W1sT + 3 * w1t, W1dT + 3 * w1t, a1s + 3 * a1st, a1d + 3 * a1st, DIN, 4, 64, W1b);
  gat(hs_bf, NS, hd_bf, ND, ei_sd, ESD, W1sT + 5 * w1t, W1dT + 5 * w1t, a1s + 5 * a1st, a1d + 5 * a1st, DIN, 4, 64, D1);
  {
    const int sTypes[4] = {0, 1, 2, 4};
    for (int i = 0; i < 4; ++i)
      k_add_bias_rows<<<cdiv_i((long)NS * HC, 256), 256, 0, stream>>>(S1, b1 + sTypes[i] * bst, (long)NS * HC, HC);
    k_add_bias_rows<<<cdiv_i((long)NW * HC, 256), 256, 0, stream>>>(W1b, b1 + 3 * bst, (long)NW * HC, HC);
    k_add_bias_rows<<<cdiv_i((long)ND * HC, 256), 256, 0, stream>>>(D1, b1 + 5 * bst, (long)ND * HC, HC);
  }
  // relu + LN -> bf16 (only consumed by conv2 GEMMs)
  k_relu_ln<<<NS, 256, 0, stream>>>(S1, nullptr, s1_bf, HC, 1);
  k_relu_ln<<<NW, 256, 0, stream>>>(W1b, nullptr, w1_bf, HC, 1);
  k_relu_ln<<<ND, 256, 0, stream>>>(D1, nullptr, d1_bf, HC, 1);

  // ---- conv2 (H=1, C=256), sentence destination only ----
  fillf(S2, (size_t)NS * HC, 0.f);
  const size_t w2t = (size_t)HC * HC;
  gat(s1_bf, NS, s1_bf, NS, ei_ss, ESS, W2sT + 0 * w2t, W2dT + 0 * w2t, a2s + 0 * 256, a2d + 0 * 256, HC, 1, 256, S2);
  gat(s1_bf, NS, s1_bf, NS, ei_sa, ESA, W2sT + 1 * w2t, W2dT + 1 * w2t, a2s + 1 * 256, a2d + 1 * 256, HC, 1, 256, S2);
  gat(w1_bf, NW, s1_bf, NS, ei_ws, EWS, W2sT + 2 * w2t, W2dT + 2 * w2t, a2s + 2 * 256, a2d + 2 * 256, HC, 1, 256, S2);
  gat(d1_bf, ND, s1_bf, NS, ei_ds, EDS, W2sT + 4 * w2t, W2dT + 4 * w2t, a2s + 4 * 256, a2d + 4 * 256, HC, 1, 256, S2);
  {
    const int sTypes[4] = {0, 1, 2, 4};
    for (int i = 0; i < 4; ++i)
      k_add_bias_rows<<<cdiv_i((long)NS * HC, 256), 256, 0, stream>>>(S2, b2 + sTypes[i] * bst, (long)NS * HC, HC);
  }

  // ---- final LN (no relu): f32 -> d_out, bf16 copy for the projection ----
  float* s2out = (float*)d_out;
  k_relu_ln<<<NS, 256, 0, stream>>>(S2, s2out, s2_bf, HC, 0);

  // ---- projection head -> d_out[NS*256 : NS*256 + NS*128) ----
  gemm_b(s2_bf, Wp1T, bp1, p1_bf, NS, 128, HC, 1);
  gemm_f(p1_bf, Wp2T, bp2, s2out + (size_t)NS * HC, NS, 128, 128, 0);
}
